// Mamba2Block_57681410785547
// MI455X (gfx1250) — matrix operand order repaired and checked by reading, not compiled, not run on hardware
//
#include <hip/hip_runtime.h>

// ---------------------------------------------------------------------------
// Mamba2-style block for MI455X (gfx1250, wave32, WMMA).
//   - dead gate branch eliminated; W_C/W_out folded into one [H,N] weight
//   - GEMMs via v_wmma_f32_16x16x32_bf16 (bf16 in, fp32 accumulate)
//   - GEMM1 (dominant) staged through LDS with global_load_async_to_lds_b128,
//     double-buffered via ASYNCcnt (s_wait_asynccnt)
//   - diagonal recurrence parallelized as a 3-pass chunked scan
// ---------------------------------------------------------------------------

typedef __bf16 bf16_t;
typedef __attribute__((ext_vector_type(16))) __bf16 v16bf;
typedef __attribute__((ext_vector_type(8)))  float  v8f;

#define BSZ    2
#define SEQ    4096
#define HDIM   1024
#define SDIM   128            // state dimension N
#define H2     2048           // 2*H
#define MROWS  (BSZ * SEQ)    // 8192
#define CHUNK  64
#define NCHUNK (SEQ / CHUNK)  // 64

// GEMM1 LDS tiling
#define BMT    128
#define BNT    128
#define KS     32
#define LPAD   40                 // row stride (elements), padded for banks
#define NKSTEP (HDIM / KS)        // 32

struct Frag { union { v16bf v; uint4 q[2]; }; };

// ---- 32x32 output macro-tile per wave, K-step 32, streaming from global ---
// A: [M,K] row-major bf16 (lda), W: [N,K] row-major bf16 (ldw) -> out = A @ W^T
__device__ __forceinline__ void gemm_core(const bf16_t* __restrict__ A, int lda,
                                          const bf16_t* __restrict__ W, int ldw,
                                          int K, int m0, int n0, int lane,
                                          v8f acc[2][2]) {
    const int r = lane & 15;
    const int h = lane >> 4;
    for (int k0 = 0; k0 < K; k0 += 32) {
        Frag a0, a1, b0, b1;
        const bf16_t* ap0 = A + (size_t)(m0 + r) * lda + k0;
        const bf16_t* ap1 = ap0 + (size_t)16 * lda;
        a0.q[0] = *(const uint4*)(ap0 + h * 8);
        a0.q[1] = *(const uint4*)(ap0 + 16 + h * 8);
        a1.q[0] = *(const uint4*)(ap1 + h * 8);
        a1.q[1] = *(const uint4*)(ap1 + 16 + h * 8);
        const bf16_t* bp0 = W + (size_t)(n0 + r) * ldw + k0 + h * 8;
        const bf16_t* bp1 = bp0 + (size_t)16 * ldw;
        b0.q[0] = *(const uint4*)(bp0);
        b0.q[1] = *(const uint4*)(bp0 + 16);
        b1.q[0] = *(const uint4*)(bp1);
        b1.q[1] = *(const uint4*)(bp1 + 16);
        __builtin_prefetch(ap0 + 32, 0, 1);
        __builtin_prefetch(bp0 + 32, 0, 1);
        acc[0][0] = __builtin_amdgcn_wmma_f32_16x16x32_bf16(false, a0.v, false, b0.v, (short)0, acc[0][0], false, false);
        acc[0][1] = __builtin_amdgcn_wmma_f32_16x16x32_bf16(false, a0.v, false, b1.v, (short)0, acc[0][1], false, false);
        acc[1][0] = __builtin_amdgcn_wmma_f32_16x16x32_bf16(false, a1.v, false, b0.v, (short)0, acc[1][0], false, false);
        acc[1][1] = __builtin_amdgcn_wmma_f32_16x16x32_bf16(false, a1.v, false, b1.v, (short)0, acc[1][1], false, false);
    }
}

// ---- GEMM1: xp = silu(x @ W_in^T + b_in)  (M=8192,N=2048,K=1024) ----------
// 128x128 block tile staged in LDS via async copies; 8 waves, each 32x64 out.
__global__ __launch_bounds__(256) void k_gemm_in_silu_lds(const bf16_t* __restrict__ A,
                                                          const bf16_t* __restrict__ W,
                                                          const float* __restrict__ bias,
                                                          bf16_t* __restrict__ out) {
    __shared__ alignas(16) bf16_t As[2][BMT][LPAD];
    __shared__ alignas(16) bf16_t Bs[2][BNT][LPAD];

    const int tid   = threadIdx.x;
    const int lane  = tid & 31;
    const int wid   = tid >> 5;
    const int waveM = wid >> 1;            // 0..3 -> 32-row band
    const int waveN = wid & 1;             // 0..1 -> 64-col band
    const int bm    = blockIdx.x >> 4;     // 2048/BNT = 16 column blocks
    const int bn    = blockIdx.x & 15;
    const int m0    = bm * BMT;
    const int n0    = bn * BNT;
    const int r     = lane & 15;
    const int h     = lane >> 4;

    // copy mapping: 256 threads cover 64 rows x (32 bf16) per round; 2 rounds
    // each for A and B -> 4 async b128 per thread per K-step.
    const int crow = tid >> 2;             // 0..63
    const int cseg = (tid & 3) * 8;        // element offset (8 bf16 = 16B)

    auto issue_async = [&](int buf, int kk) {
        uint32_t lA0 = (uint32_t)(uintptr_t)&As[buf][crow     ][cseg];
        uint32_t lA1 = (uint32_t)(uintptr_t)&As[buf][crow + 64][cseg];
        uint32_t lB0 = (uint32_t)(uintptr_t)&Bs[buf][crow     ][cseg];
        uint32_t lB1 = (uint32_t)(uintptr_t)&Bs[buf][crow + 64][cseg];
        const bf16_t* gA0 = A + (size_t)(m0 + crow     ) * HDIM + kk + cseg;
        const bf16_t* gA1 = A + (size_t)(m0 + crow + 64) * HDIM + kk + cseg;
        const bf16_t* gB0 = W + (size_t)(n0 + crow     ) * HDIM + kk + cseg;
        const bf16_t* gB1 = W + (size_t)(n0 + crow + 64) * HDIM + kk + cseg;
        asm volatile(
            "global_load_async_to_lds_b128 %0, %4, off\n\t"
            "global_load_async_to_lds_b128 %1, %5, off\n\t"
            "global_load_async_to_lds_b128 %2, %6, off\n\t"
            "global_load_async_to_lds_b128 %3, %7, off"
            :: "v"(lA0), "v"(lA1), "v"(lB0), "v"(lB1),
               "v"(gA0), "v"(gA1), "v"(gB0), "v"(gB1)
            : "memory");
    };

    v8f acc[2][4] = {};
    issue_async(0, 0);
    for (int s = 0; s < NKSTEP; ++s) {
        const int buf = s & 1;
        if (s + 1 < NKSTEP) {
            issue_async(buf ^ 1, (s + 1) * KS);                 // prefetch next
            asm volatile("s_wait_asynccnt 4" ::: "memory");     // current done
        } else {
            asm volatile("s_wait_asynccnt 0" ::: "memory");
        }
        __syncthreads();                                        // tiles visible

        Frag af[2], bfr[4];
        #pragma unroll
        for (int i = 0; i < 2; ++i) {
            const bf16_t* p = &As[buf][waveM * 32 + 16 * i + r][0];
            af[i].q[0] = *(const uint4*)(p + h * 8);
            af[i].q[1] = *(const uint4*)(p + 16 + h * 8);
        }
        #pragma unroll
        for (int j = 0; j < 4; ++j) {
            const bf16_t* p = &Bs[buf][waveN * 64 + 16 * j + r][h * 8];
            bfr[j].q[0] = *(const uint4*)(p);
            bfr[j].q[1] = *(const uint4*)(p + 16);
        }
        #pragma unroll
        for (int i = 0; i < 2; ++i)
            #pragma unroll
            for (int j = 0; j < 4; ++j)
                acc[i][j] = __builtin_amdgcn_wmma_f32_16x16x32_bf16(
                    false, af[i].v, false, bfr[j].v, (short)0, acc[i][j], false, false);
        __syncthreads();                    // buffer free for next prefetch
    }

    #pragma unroll
    for (int i = 0; i < 2; ++i)
        #pragma unroll
        for (int j = 0; j < 4; ++j)
            #pragma unroll
            for (int rr = 0; rr < 8; ++rr) {
                const int row = m0 + waveM * 32 + 16 * i + 8 * h + rr;
                const int col = n0 + waveN * 64 + 16 * j + r;
                float v = acc[i][j][rr] + bias[col];
                v = v / (1.0f + __expf(-v));                     // SiLU
                out[(size_t)row * H2 + col] = (bf16_t)v;
            }
}

// ---- GEMM2: Bt = xp @ W_B^T + b_B, fp32 out  (M=8192,N=128,K=2048) --------
__global__ __launch_bounds__(256) void k_gemm_bt(const bf16_t* __restrict__ A,
                                                 const bf16_t* __restrict__ W,
                                                 const float* __restrict__ bias,
                                                 float* __restrict__ out) {
    const int lane = threadIdx.x & 31;
    const int wave = blockIdx.x * 8 + (threadIdx.x >> 5);
    const int wcols = SDIM / 32;
    const int m0 = (wave / wcols) * 32;
    const int n0 = (wave % wcols) * 32;
    v8f acc[2][2] = {};
    gemm_core(A, H2, W, H2, H2, m0, n0, lane, acc);
    const int r = lane & 15, h = lane >> 4;
    #pragma unroll
    for (int i = 0; i < 2; ++i)
        #pragma unroll
        for (int j = 0; j < 2; ++j)
            #pragma unroll
            for (int rr = 0; rr < 8; ++rr) {
                const int row = m0 + 16 * i + 8 * h + rr;
                const int col = n0 + 16 * j + r;
                out[(size_t)row * SDIM + col] = acc[i][j][rr] + bias[col];
            }
}

// ---- GEMM3: out = hs @ Wcomb^T + bcomb + x  (M=8192,N=1024,K=128) ---------
__global__ __launch_bounds__(256) void k_gemm_out(const bf16_t* __restrict__ A,
                                                  const bf16_t* __restrict__ W,
                                                  const float* __restrict__ bias,
                                                  const float* __restrict__ xres,
                                                  float* __restrict__ out) {
    const int lane = threadIdx.x & 31;
    const int wave = blockIdx.x * 8 + (threadIdx.x >> 5);
    const int wcols = HDIM / 32;
    const int m0 = (wave / wcols) * 32;
    const int n0 = (wave % wcols) * 32;
    v8f acc[2][2] = {};
    gemm_core(A, SDIM, W, SDIM, SDIM, m0, n0, lane, acc);
    const int r = lane & 15, h = lane >> 4;
    #pragma unroll
    for (int i = 0; i < 2; ++i)
        #pragma unroll
        for (int j = 0; j < 2; ++j)
            #pragma unroll
            for (int rr = 0; rr < 8; ++rr) {
                const int row = m0 + 16 * i + 8 * h + rr;
                const int col = n0 + 16 * j + r;
                const size_t idx = (size_t)row * HDIM + col;
                out[idx] = acc[i][j][rr] + bias[col] + xres[idx];
            }
}

// ---- helpers ---------------------------------------------------------------
__global__ void k_cast_bf16(const float* __restrict__ src, bf16_t* __restrict__ dst, int n) {
    for (int i = blockIdx.x * blockDim.x + threadIdx.x; i < n; i += gridDim.x * blockDim.x)
        dst[i] = (bf16_t)src[i];
}

// W_comb[o,n] = sum_h W_out[o,h] * W_C[h,n]  (fp32 accumulate, bf16 store)
__global__ __launch_bounds__(128) void k_wcomb(const float* __restrict__ Wout,
                                               const float* __restrict__ Wc,
                                               bf16_t* __restrict__ wbf) {
    const int o = blockIdx.x;
    const int n = threadIdx.x;
    float s = 0.f;
    for (int hh = 0; hh < HDIM; ++hh)
        s = fmaf(Wout[(size_t)o * HDIM + hh], Wc[(size_t)hh * SDIM + n], s);
    wbf[o * SDIM + n] = (bf16_t)s;
}

// b_comb[o] = b_out[o] + sum_h W_out[o,h] * b_C[h]
__global__ void k_bcomb(const float* __restrict__ Wout, const float* __restrict__ bC,
                        const float* __restrict__ bout, float* __restrict__ bcomb) {
    const int o = blockIdx.x * blockDim.x + threadIdx.x;
    if (o >= HDIM) return;
    float s = bout[o];
    for (int hh = 0; hh < HDIM; ++hh)
        s = fmaf(Wout[(size_t)o * HDIM + hh], bC[hh], s);
    bcomb[o] = s;
}

// a[n] = exp(A_log)+A_bias ; aL[n] = a^CHUNK
__global__ void k_avec(const float* __restrict__ A_log, const float* __restrict__ A_bias,
                       float* __restrict__ a, float* __restrict__ aL) {
    const int n = threadIdx.x;
    if (n < SDIM) {
        const float av = __expf(A_log[n]) + A_bias[n];
        a[n] = av;
        float p = av;
        #pragma unroll
        for (int i = 0; i < 6; ++i) p *= p;   // a^64
        aL[n] = p;
    }
}

// pass1: per-chunk local scan from 0 -> chunk-final state
__global__ __launch_bounds__(256) void k_scan1(const float* __restrict__ Bt,
                                               const float* __restrict__ a,
                                               float* __restrict__ chunkf) {
    const int tid = blockIdx.x * 256 + threadIdx.x;   // (b*NCHUNK + c)*SDIM + n
    const int n = tid & (SDIM - 1);
    const int c = (tid >> 7) & (NCHUNK - 1);
    const int b = tid >> 13;
    const float an = a[n];
    const float* p = Bt + ((size_t)(b * SEQ + c * CHUNK)) * SDIM + n;
    float f = 0.f;
    for (int t = 0; t < CHUNK; ++t) f = fmaf(an, f, p[(size_t)t * SDIM]);
    chunkf[tid] = f;
}

// pass2: sequential carry across chunks (tiny)
__global__ void k_scan2(const float* __restrict__ chunkf, const float* __restrict__ aL,
                        float* __restrict__ Hinit) {
    const int tid = threadIdx.x;          // 256 = BSZ*SDIM
    const int n = tid & (SDIM - 1);
    const int b = tid >> 7;
    const float al = aL[n];
    float hcur = 0.f;
    for (int c = 0; c < NCHUNK; ++c) {
        const int idx = (b * NCHUNK + c) * SDIM + n;
        Hinit[idx] = hcur;
        hcur = fmaf(al, hcur, chunkf[idx]);
    }
}

// pass3: replay with correct init state, emit hs as bf16
__global__ __launch_bounds__(256) void k_scan3(const float* __restrict__ Bt,
                                               const float* __restrict__ a,
                                               const float* __restrict__ Hinit,
                                               bf16_t* __restrict__ hs) {
    const int tid = blockIdx.x * 256 + threadIdx.x;
    const int n = tid & (SDIM - 1);
    const int c = (tid >> 7) & (NCHUNK - 1);
    const int b = tid >> 13;
    const float an = a[n];
    const size_t base = ((size_t)(b * SEQ + c * CHUNK)) * SDIM + n;
    float hv = Hinit[tid];
    for (int t = 0; t < CHUNK; ++t) {
        hv = fmaf(an, hv, Bt[base + (size_t)t * SDIM]);
        hs[base + (size_t)t * SDIM] = (bf16_t)hv;
    }
}

// in-place LayerNorm, one wave32 per row of 1024
__global__ __launch_bounds__(256) void k_ln(float* __restrict__ y,
                                            const float* __restrict__ g,
                                            const float* __restrict__ bb) {
    const int lane = threadIdx.x & 31;
    const int row = blockIdx.x * 8 + (threadIdx.x >> 5);
    float* p = y + (size_t)row * HDIM;
    float vals[32];
    float s = 0.f, sq = 0.f;
    #pragma unroll
    for (int i = 0; i < 32; ++i) {
        const float v = p[lane + 32 * i];
        vals[i] = v; s += v; sq += v * v;
    }
    #pragma unroll
    for (int off = 16; off; off >>= 1) {
        s  += __shfl_xor(s,  off, 32);
        sq += __shfl_xor(sq, off, 32);
    }
    const float mu  = s * (1.0f / HDIM);
    const float var = sq * (1.0f / HDIM) - mu * mu;
    const float rs  = rsqrtf(var + 1e-5f);
    #pragma unroll
    for (int i = 0; i < 32; ++i) {
        const int col = lane + 32 * i;
        p[col] = (vals[i] - mu) * rs * g[col] + bb[col];
    }
}

// ---------------------------------------------------------------------------
extern "C" void kernel_launch(void* const* d_in, const int* in_sizes, int n_in,
                              void* d_out, int out_size, void* d_ws, size_t ws_size,
                              hipStream_t stream) {
    const float* x      = (const float*)d_in[0];
    const float* W_in   = (const float*)d_in[1];
    const float* b_in   = (const float*)d_in[2];
    const float* A_log  = (const float*)d_in[3];
    const float* A_bias = (const float*)d_in[4];
    const float* W_B    = (const float*)d_in[5];
    const float* b_B    = (const float*)d_in[6];
    // d_in[7..10] = Wg1,bg1,Wg2,bg2 : dead code in the reference output, skipped
    const float* W_C    = (const float*)d_in[11];
    const float* b_C    = (const float*)d_in[12];
    const float* W_out  = (const float*)d_in[13];
    const float* b_out  = (const float*)d_in[14];
    const float* ln_g   = (const float*)d_in[15];
    const float* ln_b   = (const float*)d_in[16];
    float* out = (float*)d_out;

    char* ws = (char*)d_ws;
    size_t off = 0;
    auto alloc = [&](size_t bytes) -> void* {
        void* p = ws + off;
        off = (off + bytes + 255) & ~(size_t)255;
        return p;
    };
    bf16_t* xbf    = (bf16_t*)alloc((size_t)MROWS * HDIM * 2);   // 16.8 MB
    bf16_t* winbf  = (bf16_t*)alloc((size_t)H2 * HDIM * 2);      //  4.2 MB
    bf16_t* wbbf   = (bf16_t*)alloc((size_t)SDIM * H2 * 2);      //  0.5 MB
    bf16_t* wcomb  = (bf16_t*)alloc((size_t)HDIM * SDIM * 2);    //  0.3 MB
    float*  bcomb  = (float*) alloc((size_t)HDIM * 4);
    float*  avec   = (float*) alloc((size_t)SDIM * 4);
    float*  aLvec  = (float*) alloc((size_t)SDIM * 4);
    bf16_t* xp     = (bf16_t*)alloc((size_t)MROWS * H2 * 2);     // 33.6 MB
    float*  Bt     = (float*) alloc((size_t)MROWS * SDIM * 4);   //  4.2 MB
    float*  chunkf = (float*) alloc((size_t)BSZ * NCHUNK * SDIM * 4);
    float*  Hinit  = (float*) alloc((size_t)BSZ * NCHUNK * SDIM * 4);
    bf16_t* hs     = (bf16_t*)alloc((size_t)MROWS * SDIM * 2);   //  2.1 MB

    // --- precompute / casts ---
    k_cast_bf16<<<2048, 256, 0, stream>>>(x,    xbf,   MROWS * HDIM);
    k_cast_bf16<<<1024, 256, 0, stream>>>(W_in, winbf, H2 * HDIM);
    k_cast_bf16<<<256,  256, 0, stream>>>(W_B,  wbbf,  SDIM * H2);
    k_wcomb<<<HDIM, SDIM, 0, stream>>>(W_out, W_C, wcomb);
    k_bcomb<<<4, 256, 0, stream>>>(W_out, b_C, b_out, bcomb);
    k_avec<<<1, 128, 0, stream>>>(A_log, A_bias, avec, aLvec);

    // --- xp = silu(x @ W_in^T + b_in) : LDS-tiled, async double-buffered ---
    k_gemm_in_silu_lds<<<(MROWS / BMT) * (H2 / BNT), 256, 0, stream>>>(xbf, winbf, b_in, xp);

    // --- Bt = xp @ W_B^T + b_B ---
    k_gemm_bt<<<(MROWS / 32) * (SDIM / 32) / 8, 256, 0, stream>>>(xp, wbbf, b_B, Bt);

    // --- chunked parallel scan: h_t = a*h_{t-1} + Bt ---
    k_scan1<<<(BSZ * NCHUNK * SDIM) / 256, 256, 0, stream>>>(Bt, avec, chunkf);
    k_scan2<<<1, BSZ * SDIM, 0, stream>>>(chunkf, aLvec, Hinit);
    k_scan3<<<(BSZ * NCHUNK * SDIM) / 256, 256, 0, stream>>>(Bt, avec, Hinit, hs);

    // --- out = hs @ (W_out@W_C)^T + (W_out@b_C + b_out) + x ---
    k_gemm_out<<<(MROWS / 32) * (HDIM / 32) / 8, 256, 0, stream>>>(hs, wcomb, bcomb, x, out);

    // --- in-place LayerNorm ---
    k_ln<<<MROWS / 8, 256, 0, stream>>>(out, ln_g, ln_b);
}
